// bi_Mlp_17231408792274
// MI455X (gfx1250) — compile-verified, matrix operand order checked by reading, not run on hardware
//
#include <hip/hip_runtime.h>
#include <cmath>

typedef __attribute__((ext_vector_type(16))) _Float16 v16h;
typedef __attribute__((ext_vector_type(8)))  _Float16 v8h;
typedef __attribute__((ext_vector_type(8)))  float    v8f;
typedef __attribute__((ext_vector_type(4)))  int      v4i;

#define BM 64
#define BN 128
#define BK 64
#define PAD 8
#define LDK (BK + PAD)

#if __has_builtin(__builtin_amdgcn_global_load_async_to_lds_b128)
#define HAVE_ASYNC_LDS 1
typedef __attribute__((address_space(1))) v4i* gas_b128_t;
typedef __attribute__((address_space(3))) v4i* las_b128_t;
#else
#define HAVE_ASYNC_LDS 0
#endif

// 16-byte global -> LDS copy; async DMA (ASYNCcnt) when available.
__device__ __forceinline__ void copy16_g2l(const _Float16* __restrict__ g,
                                           _Float16* __restrict__ l) {
#if HAVE_ASYNC_LDS
  __builtin_amdgcn_global_load_async_to_lds_b128((gas_b128_t)g, (las_b128_t)l,
                                                 0, 0);
#else
  *(v8h*)l = *(const v8h*)g;
#endif
}

__device__ __forceinline__ void wait_async_copies() {
#if HAVE_ASYNC_LDS
#if __has_builtin(__builtin_amdgcn_s_wait_asynccnt)
  __builtin_amdgcn_s_wait_asynccnt(0);
#endif
#endif
}

// ---------------------------------------------------------------------------
// Per-row binarization: alpha[r] = mean(|w[r,:]|), ws[r,c] = sign(w[r,c]) (f16)
// ---------------------------------------------------------------------------
__global__ __launch_bounds__(256) void binarize_kernel(
    const float* __restrict__ w, _Float16* __restrict__ ws,
    float* __restrict__ alpha, int cols) {
  const int row = blockIdx.x;
  const float* wr = w + (size_t)row * cols;
  _Float16* wsr = ws + (size_t)row * cols;
  float s = 0.0f;
  for (int c = threadIdx.x; c < cols; c += blockDim.x) {
    float v = wr[c];
    s += fabsf(v);
    wsr[c] = (_Float16)((v > 0.0f) ? 1.0f : ((v < 0.0f) ? -1.0f : 0.0f));
  }
  __shared__ float red[256];
  red[threadIdx.x] = s;
  __syncthreads();
  for (int off = 128; off > 0; off >>= 1) {
    if (threadIdx.x < off) red[threadIdx.x] += red[threadIdx.x + off];
    __syncthreads();
  }
  if (threadIdx.x == 0) alpha[row] = red[0] / (float)cols;
}

// ---------------------------------------------------------------------------
// WMMA GEMM:  Out[M, Ntot] = epilogue( A[M,K] * B[Ntot,K]^T * alpha[n] + bias[n] )
// A: f32 (GEMM1, x) or f16 (GEMM2, h).  B: f16 sign weights, row-major [Ntot,K].
// Block tile 64x128, K-tile 64, 8 wave32 waves; each wave: 2x2 WMMA 16x16 tiles.
// Pure-f16 tiles staged via async global->LDS DMA; f32 A converted through VGPRs.
// ---------------------------------------------------------------------------
template <bool A_F32, bool GELU_EPI>
__global__ __launch_bounds__(256) void gemm_wmma(
    const void* __restrict__ Aptr, const _Float16* __restrict__ B,
    const float* __restrict__ alpha, const float* __restrict__ bias,
    void* __restrict__ Out, int K, int Ntot) {
  __shared__ _Float16 sA[2][BM][LDK];
  __shared__ _Float16 sB[2][BN][LDK];

  const int tid  = threadIdx.x;
  const int lane = tid & 31;
  const int wave = tid >> 5;
  const int lo   = lane & 15;
  const int hi   = lane >> 4;
  const int wm   = (wave & 1) * 32;   // wave row offset within block tile
  const int wn   = (wave >> 1) * 32;  // wave col offset within block tile

  const int m0 = blockIdx.x * BM;
  const int n0 = blockIdx.y * BN;

  // global -> LDS staging assignments (256 threads)
  const int rA = tid >> 2;            // 0..63, 4 threads per A row (16 elems each)
  const int cA = (tid & 3) * 16;
  const int rB = tid >> 1;            // 0..127, 2 threads per B row (32 elems each)
  const int cB = (tid & 1) * 32;

  v8f acc[2][2];
#pragma unroll
  for (int i = 0; i < 2; ++i)
#pragma unroll
    for (int j = 0; j < 2; ++j) acc[i][j] = {};

  auto load_stage = [&](int s, int k0) {
    if constexpr (A_F32) {
      const float* Af = (const float*)Aptr;
      const float4* src = (const float4*)(Af + (size_t)(m0 + rA) * K + k0 + cA);
#pragma unroll
      for (int j = 0; j < 4; ++j) {
        float4 v = src[j];
        _Float16* dst = &sA[s][rA][cA + j * 4];
        dst[0] = (_Float16)v.x; dst[1] = (_Float16)v.y;
        dst[2] = (_Float16)v.z; dst[3] = (_Float16)v.w;
      }
    } else {
      const _Float16* Ah = (const _Float16*)Aptr;
      const _Float16* src = Ah + (size_t)(m0 + rA) * K + k0 + cA;
#pragma unroll
      for (int j = 0; j < 2; ++j)
        copy16_g2l(src + j * 8, &sA[s][rA][cA + j * 8]);
    }
    const _Float16* bs = B + (size_t)(n0 + rB) * K + k0 + cB;
#pragma unroll
    for (int j = 0; j < 4; ++j)
      copy16_g2l(bs + j * 8, &sB[s][rB][cB + j * 8]);
  };

  // ISA 16-bit A/B fragment layout: lane(0..15)=row lo, chunks at k+hi*8 and k+16+hi*8
  auto frag = [&](const _Float16* rowp, int kk) -> v16h {
    v8h c0 = *(const v8h*)(rowp + kk + hi * 8);
    v8h c1 = *(const v8h*)(rowp + kk + 16 + hi * 8);
    return __builtin_shufflevector(c0, c1, 0, 1, 2, 3, 4, 5, 6, 7, 8, 9, 10, 11,
                                   12, 13, 14, 15);
  };

  load_stage(0, 0);
  wait_async_copies();
  __syncthreads();

  const int nk = K / BK;
  for (int kb = 0; kb < nk; ++kb) {
    const int s = kb & 1;
    if (kb + 1 < nk) load_stage(s ^ 1, (kb + 1) * BK);
#pragma unroll
    for (int kk = 0; kk < BK; kk += 32) {
      v16h af[2], bf[2];
#pragma unroll
      for (int i = 0; i < 2; ++i) af[i] = frag(&sA[s][wm + i * 16 + lo][0], kk);
#pragma unroll
      for (int j = 0; j < 2; ++j) bf[j] = frag(&sB[s][wn + j * 16 + lo][0], kk);
#pragma unroll
      for (int i = 0; i < 2; ++i)
#pragma unroll
        for (int j = 0; j < 2; ++j)
          acc[i][j] = __builtin_amdgcn_wmma_f32_16x16x32_f16(
              false, af[i], false, bf[j], (short)0, acc[i][j], false, false);
    }
    wait_async_copies();
    __syncthreads();
  }

  // Epilogue: C/D layout -> lane holds col n = lo, VGPR v holds row v + 8*hi
#pragma unroll
  for (int j = 0; j < 2; ++j) {
    const int n = n0 + wn + j * 16 + lo;
    const float al = alpha[n];
    const float bi = bias[n];
#pragma unroll
    for (int i = 0; i < 2; ++i) {
      const int mbase = m0 + wm + i * 16 + hi * 8;
#pragma unroll
      for (int v = 0; v < 8; ++v) {
        float val = acc[i][j][v] * al + bi;
        if constexpr (GELU_EPI) {
          val = 0.5f * val * (1.0f + erff(val * 0.70710678118654752f));
          val = fminf(fmaxf(val, -10.0f), 10.0f);
          ((_Float16*)Out)[(size_t)(mbase + v) * Ntot + n] = (_Float16)val;
        } else {
          ((float*)Out)[(size_t)(mbase + v) * Ntot + n] = val;
        }
      }
    }
  }
}

// ---------------------------------------------------------------------------
extern "C" void kernel_launch(void* const* d_in, const int* in_sizes, int n_in,
                              void* d_out, int out_size, void* d_ws,
                              size_t ws_size, hipStream_t stream) {
  (void)in_sizes; (void)n_in; (void)out_size; (void)ws_size;
  const float* x  = (const float*)d_in[0];  // [64,197,768]
  const float* w1 = (const float*)d_in[1];  // [3072,768]
  const float* b1 = (const float*)d_in[2];  // [3072]
  const float* w2 = (const float*)d_in[3];  // [768,3072]
  const float* b2 = (const float*)d_in[4];  // [768]
  float* out = (float*)d_out;               // [64,197,768]

  const int C = 768, H = 3072;
  const int M = 64 * 197;  // 12608 = 197 * 64

  char* ws = (char*)d_ws;
  size_t off = 0;
  auto alloc = [&](size_t bytes) -> void* {
    void* p = ws + off;
    off = (off + bytes + 255) & ~(size_t)255;
    return p;
  };
  _Float16* w1s   = (_Float16*)alloc((size_t)H * C * sizeof(_Float16));
  _Float16* w2s   = (_Float16*)alloc((size_t)C * H * sizeof(_Float16));
  float*    al1   = (float*)alloc((size_t)H * sizeof(float));
  float*    al2   = (float*)alloc((size_t)C * sizeof(float));
  _Float16* hbuf  = (_Float16*)alloc((size_t)M * H * sizeof(_Float16));

  binarize_kernel<<<H, 256, 0, stream>>>(w1, w1s, al1, C);
  binarize_kernel<<<C, 256, 0, stream>>>(w2, w2s, al2, H);

  // GEMM1: h = clip(gelu(x @ sign(w1)^T * a1 + b1))  -> f16 [M,H]
  gemm_wmma<true, true><<<dim3(M / BM, H / BN), 256, 0, stream>>>(
      x, w1s, al1, b1, hbuf, C, H);
  // GEMM2: out = h @ sign(w2)^T * a2 + b2            -> f32 [M,C]
  gemm_wmma<false, false><<<dim3(M / BM, C / BN), 256, 0, stream>>>(
      hbuf, w2s, al2, b2, out, H, C);
}
